// AplhaMixerAttentionHeads_35725537968576
// MI455X (gfx1250) — compile-verified, matrix operand order checked by reading, not run on hardware
//
#include <hip/hip_runtime.h>
#include <hip/hip_bf16.h>

// ---------------- problem constants ----------------
#define B_   2
#define S_   1024
#define FIN  768
#define E_   768
#define H_   12
#define TOK  (B_*S_)      // 2048
#define MIN_POS 1e-6f
#define EPSF    1e-20f

typedef _Float16 f16;
typedef __attribute__((ext_vector_type(16))) _Float16 v16h;
typedef __attribute__((ext_vector_type(8)))  _Float16 v8h;
typedef __attribute__((ext_vector_type(4)))  _Float16 v4h;
typedef __attribute__((ext_vector_type(8)))  float    v8f;

// ---------------- WMMA helper (wave32, 16x16x32 f16 -> f32) ----------------
__device__ __forceinline__ v8f wmma32(v16h a, v16h b, v8f c) {
  return __builtin_amdgcn_wmma_f32_16x16x32_f16(false, a, false, b, (short)0, c,
                                                false, false);
}

// Fragment from a ROW-MAJOR matrix where lane (L&15) indexes the row:
//  - A operand from M x K row-major
//  - B operand from N x K row-major (i.e. B^T stored row-major)
// K chunks {kb..kb+7, kb+16..kb+23}, kb = 8*(L>=16). Two 16-byte loads.
__device__ __forceinline__ v16h frag_rm_h(const f16* base, int ld, int lane) {
  const int r  = lane & 15;
  const int kb = (lane & 16) >> 1;   // 0 or 8
  const f16* q = base + (long)r * ld + kb;
  v8h lo = *(const v8h*)(q);
  v8h hi = *(const v8h*)(q + 16);
  return __builtin_shufflevector(lo, hi, 0,1,2,3,4,5,6,7,8,9,10,11,12,13,14,15);
}

// Same gather from an f32 source (convert on load; used on LDS tiles only).
__device__ __forceinline__ v16h frag_rm_f32(const float* base, int ld, int lane) {
  const int r  = lane & 15;
  const int kb = (lane & 16) >> 1;
  const float* q = base + (long)r * ld + kb;
  v16h a;
#pragma unroll
  for (int e = 0; e < 8; ++e) { a[e] = (f16)q[e]; a[e+8] = (f16)q[e+16]; }
  return a;
}

// ---------------- K0: normalize NMF dictionary -> Wn (FHxDH), WnT (DHxFH) f16
__global__ void k_prep_w(const float* __restrict__ nw, f16* __restrict__ Wn,
                         f16* __restrict__ WnT) {
  int r = threadIdx.x;
  if (r < 64) {
    float s = 0.f;
    for (int d = 0; d < 64; ++d) s += nw[r*64 + d];
    float inv = 1.f / fmaxf(s, EPSF);
    for (int d = 0; d < 64; ++d) {
      float wv = nw[r*64 + d] * inv;
      Wn [r*64 + d] = (f16)wv;
      WnT[d*64 + r] = (f16)wv;
    }
  }
}

// ---------------- K0b: bulk f32 -> f16 conversion (4-wide)
__global__ void k_cvt_f16(const float* __restrict__ s, f16* __restrict__ d, long n4) {
  long i = (long)blockIdx.x * blockDim.x + threadIdx.x;
  if (i < n4) {
    const float4 v = ((const float4*)s)[i];
    v4h h;
    h[0] = (f16)v.x; h[1] = (f16)v.y; h[2] = (f16)v.z; h[3] = (f16)v.w;
    ((v4h*)d)[i] = h;
  }
}

// ---------------- generic GEMM: C[MxN] = A[MxK] * Bt[NxK]^T (+bias, opt clamp)
// 8 waves/block; each wave one 16x64 tile; ping-pong pipelined K loop.
// Requires K % 64 == 0.
__global__ void __launch_bounds__(256)
k_gemm16x64(const f16* __restrict__ A, const f16* __restrict__ Bt,
            const float* __restrict__ bias, float* __restrict__ C,
            int K, int N, int doClamp) {
  const int lane = threadIdx.x & 31, w = threadIdx.x >> 5;
  const int t  = blockIdx.x * 8 + w;
  const int nT = N / 64;
  const int tm = t / nT, tn = t % nT;
  const f16* arow = A  + (long)tm * 16 * K;
  const f16* brow = Bt + (long)tn * 64 * K;
  v8f acc[4] = {};
  v16h aF0 = frag_rm_h(arow, K, lane);
  v16h bF0[4];
#pragma unroll
  for (int nt = 0; nt < 4; ++nt) bF0[nt] = frag_rm_h(brow + (long)nt*16*K, K, lane);
  for (int k0 = 0; k0 < K; k0 += 64) {
    const int kA = k0 + 32;                        // always < K
    v16h aF1 = frag_rm_h(arow + kA, K, lane);
    v16h bF1[4];
#pragma unroll
    for (int nt = 0; nt < 4; ++nt)
      bF1[nt] = frag_rm_h(brow + (long)nt*16*K + kA, K, lane);
#pragma unroll
    for (int nt = 0; nt < 4; ++nt) acc[nt] = wmma32(aF0, bF0[nt], acc[nt]);
    const int kB = (k0 + 64 < K) ? k0 + 64 : 0;    // clamped (dead on last iter)
    aF0 = frag_rm_h(arow + kB, K, lane);
#pragma unroll
    for (int nt = 0; nt < 4; ++nt)
      bF0[nt] = frag_rm_h(brow + (long)nt*16*K + kB, K, lane);
#pragma unroll
    for (int nt = 0; nt < 4; ++nt) acc[nt] = wmma32(aF1, bF1[nt], acc[nt]);
  }
  const int rb = tm*16 + ((lane & 16) >> 1);
#pragma unroll
  for (int nt = 0; nt < 4; ++nt) {
    int n = tn*64 + nt*16 + (lane & 15);
    float bs = bias[n];
#pragma unroll
    for (int j = 0; j < 8; ++j) {
      float v = acc[nt][j] + bs;
      C[(long)(rb + j) * N + n] = doClamp ? fmaxf(v, MIN_POS) : v;
    }
  }
}

// ---------------- K2: inp = l1norm over last 64 (per token-head)
__global__ void k_l1norm(const float* __restrict__ xe, float* __restrict__ inp) {
  int th = blockIdx.x * blockDim.x + threadIdx.x;   // [0, TOK*H)
  if (th >= TOK * H_) return;
  const float* p = xe + (long)th * 64;
  float s = 0.f;
  for (int i = 0; i < 64; ++i) s += p[i];           // already positive
  float inv = 1.f / fmaxf(s, EPSF);
  float* q = inp + (long)th * 64;
  for (int i = 0; i < 64; ++i) q[i] = p[i] * inv;
}

// ---------------- K3: NNMF multiplicative updates (3 iters), 16-token tile/head.
// Outputs: hT   [B,H,64,S] f16  (h transposed; B^T operand for the mixes)
//          hrec [B,H,S,64] f16  (rec*inp o-major; B^T operand for alpha update)
__global__ void __launch_bounds__(128)
k_nnmf(const float* __restrict__ inp, const f16* __restrict__ Wn,
       const f16* __restrict__ WnT, f16* __restrict__ hT,
       f16* __restrict__ hrec) {
  __shared__ alignas(16) float sI[4][16*64];
  __shared__ alignas(16) float sH[4][16*64];
  __shared__ alignas(16) float sT[4][16*64];
  const int lane = threadIdx.x & 31, w = threadIdx.x >> 5;
  const int tb   = blockIdx.x;              // token block (0..127)
  const int head = blockIdx.y * 4 + w;      // 0..11
  const int tok0 = tb * 16;
  const int b    = tok0 / S_;
  const int ls0  = tok0 - b * S_;
  const long bh  = (long)b * H_ + head;

  for (int i = lane; i < 16*64; i += 32) {
    int r = i >> 6, f = i & 63;
    sI[w][i] = inp[((long)(tok0 + r) * H_ + head) * 64 + f];
    sH[w][i] = 1.0f / 64.0f;
  }
  __syncthreads();

  for (int it = 0; it < 3; ++it) {
    // --- rec_raw = h @ Wn  (B^T = WnT row-major) -> sT
    v8f ct[4] = {};
#pragma unroll
    for (int kk = 0; kk < 2; ++kk) {
      v16h a = frag_rm_f32(&sH[w][kk*32], 64, lane);
#pragma unroll
      for (int nt = 0; nt < 4; ++nt) {
        v16h bb = frag_rm_h(WnT + (nt*16)*64 + kk*32, 64, lane);
        ct[nt] = wmma32(a, bb, ct[nt]);
      }
    }
#pragma unroll
    for (int nt = 0; nt < 4; ++nt) {
      int col = nt*16 + (lane & 15), r0 = (lane & 16) >> 1;
#pragma unroll
      for (int j = 0; j < 8; ++j) sT[w][(r0 + j)*64 + col] = ct[nt][j];
    }
    __syncthreads();

    // --- rec = l1norm(clamp(rec_raw)); last iter emits rec*inp; sT <- inp/rec
    if (lane < 16) {
      float s = 0.f;
      for (int f = 0; f < 64; ++f) s += fmaxf(sT[w][lane*64 + f], MIN_POS);
      float invs = 1.f / fmaxf(s, EPSF);
      if (it == 2) {
        for (int f = 0; f < 64; ++f) {
          float rec = fmaxf(sT[w][lane*64 + f], MIN_POS) * invs;
          hrec[(bh*S_ + ls0 + lane)*64 + f] = (f16)(rec * sI[w][lane*64 + f]);
        }
      }
      for (int f = 0; f < 64; ++f) {
        float rec = fmaxf(sT[w][lane*64 + f], MIN_POS) * invs;
        sT[w][lane*64 + f] = sI[w][lane*64 + f] / rec;
      }
    }
    __syncthreads();

    // --- h = clamp(h * ((inp/rec) @ WnT), MIN_POS)  (B^T = Wn row-major)
    v8f cu[4] = {};
#pragma unroll
    for (int kk = 0; kk < 2; ++kk) {
      v16h a = frag_rm_f32(&sT[w][kk*32], 64, lane);
#pragma unroll
      for (int nt = 0; nt < 4; ++nt) {
        v16h bb = frag_rm_h(Wn + (nt*16)*64 + kk*32, 64, lane);
        cu[nt] = wmma32(a, bb, cu[nt]);
      }
    }
#pragma unroll
    for (int nt = 0; nt < 4; ++nt) {
      int col = nt*16 + (lane & 15), r0 = (lane & 16) >> 1;
#pragma unroll
      for (int j = 0; j < 8; ++j) {
        int idx = (r0 + j)*64 + col;
        sH[w][idx] = fmaxf(sH[w][idx] * cu[nt][j], MIN_POS);
      }
    }
    __syncthreads();

    // --- h row l1norm
    if (lane < 16) {
      float s = 0.f;
      for (int f = 0; f < 64; ++f) s += sH[w][lane*64 + f];
      float invs = 1.f / fmaxf(s, EPSF);
      for (int f = 0; f < 64; ++f) sH[w][lane*64 + f] *= invs;
    }
    __syncthreads();
  }

  // store h transposed per head: hT[bh][f][token]
  for (int i = lane; i < 16*64; i += 32) {
    int r = i >> 6, f = i & 63;
    hT[(bh*64 + f)*S_ + ls0 + r] = (f16)sH[w][i];
  }
}

// ---------------- K4: AT = 1.0 (f16), 16B stores
__global__ void k_fill_ones(f16* __restrict__ AT, long n8) {
  long i = (long)blockIdx.x * blockDim.x + threadIdx.x;
  if (i < n8)
    ((uint4*)AT)[i] = make_uint4(0x3C003C00u, 0x3C003C00u, 0x3C003C00u, 0x3C003C00u);
}

// ---------------- K5: h_mixed[i,:] = AT[i,:] @ h ; arec_inv = 1/(h_mixed@Wn + eps)
// grid (S/64, H, B), block 128; wave -> one 16-row i-block; ping-pong K loop.
__global__ void __launch_bounds__(128)
k_mix_arec(const f16* __restrict__ AT, const f16* __restrict__ hT,
           const f16* __restrict__ WnT, f16* __restrict__ areci) {
  __shared__ alignas(16) f16 sM[4][16*64];    // per-wave h_mixed tile
  const int lane = threadIdx.x & 31, w = threadIdx.x >> 5;
  const int bh = blockIdx.z * H_ + blockIdx.y;
  const f16* hTh  = hT + (long)bh * 64 * S_;
  const int ibase = (blockIdx.x * 4 + w) * 16;
  const f16* Arow = AT + ((long)bh * S_ + ibase) * S_;

  v8f acc[4] = {};
  v16h aF0 = frag_rm_h(Arow, S_, lane);
  v16h bF0[4];
#pragma unroll
  for (int nt = 0; nt < 4; ++nt) bF0[nt] = frag_rm_h(hTh + (long)(nt*16)*S_, S_, lane);
  for (int o0 = 0; o0 < S_; o0 += 64) {
    const int oA = o0 + 32;                        // always < S_
    __builtin_prefetch(Arow + (long)(lane & 15) * S_ + oA + 32, 0, 1);
    v16h aF1 = frag_rm_h(Arow + oA, S_, lane);
    v16h bF1[4];
#pragma unroll
    for (int nt = 0; nt < 4; ++nt)
      bF1[nt] = frag_rm_h(hTh + (long)(nt*16)*S_ + oA, S_, lane);
#pragma unroll
    for (int nt = 0; nt < 4; ++nt) acc[nt] = wmma32(aF0, bF0[nt], acc[nt]);
    const int oB = (o0 + 64 < S_) ? o0 + 64 : 0;   // clamped (dead on last iter)
    aF0 = frag_rm_h(Arow + oB, S_, lane);
#pragma unroll
    for (int nt = 0; nt < 4; ++nt)
      bF0[nt] = frag_rm_h(hTh + (long)(nt*16)*S_ + oB, S_, lane);
#pragma unroll
    for (int nt = 0; nt < 4; ++nt) acc[nt] = wmma32(aF1, bF1[nt], acc[nt]);
  }
#pragma unroll
  for (int nt = 0; nt < 4; ++nt) {
    int col = nt*16 + (lane & 15), r0 = (lane & 16) >> 1;
#pragma unroll
    for (int j = 0; j < 8; ++j) sM[w][(r0 + j)*64 + col] = (f16)acc[nt][j];
  }
  __syncthreads();
  // arec = h_mixed @ Wn (B^T = WnT row-major); store 1/(arec+eps)
#pragma unroll
  for (int nt = 0; nt < 4; ++nt) {
    v8f c = {};
#pragma unroll
    for (int kk = 0; kk < 2; ++kk) {
      v16h a2 = frag_rm_h(&sM[w][kk*32], 64, lane);
      v16h b2 = frag_rm_h(WnT + (nt*16)*64 + kk*32, 64, lane);
      c = wmma32(a2, b2, c);
    }
    int col = nt*16 + (lane & 15), r0 = (lane & 16) >> 1;
#pragma unroll
    for (int j = 0; j < 8; ++j)
      areci[((long)bh*S_ + ibase + r0 + j)*64 + col] = (f16)(1.0f / (c[j] + EPSF));
  }
}

// ---------------- K6: AT[i,o] *= sum_f areci[i,f] * hrec[o,f]  (RMW)
// wave -> 16x64 tile (A-fragment reused by 4 WMMAs), K=64.
__global__ void __launch_bounds__(256)
k_update(f16* __restrict__ AT, const f16* __restrict__ areci,
         const f16* __restrict__ hrec) {
  const int lane = threadIdx.x & 31, w = threadIdx.x >> 5;
  const int bh = blockIdx.z * H_ + blockIdx.y;
  const int t  = blockIdx.x * 8 + w;              // 0..1023
  const int ibase = (t >> 4) * 16;                // 64 i-blocks
  const int ob64  = (t & 15) * 64;                // 16 o-blocks of 64
  const f16* Arow = areci + ((long)bh * S_ + ibase) * 64;
  const f16* Brow = hrec  + ((long)bh * S_ + ob64) * 64;   // B^T: row o, col f
  f16* Ap = AT + (long)bh * S_ * S_;
  __builtin_prefetch(Ap + (long)(ibase + (lane & 15)) * S_ + ob64, 1, 1);
  v8f acc[4] = {};
#pragma unroll
  for (int kk = 0; kk < 2; ++kk) {
    v16h a = frag_rm_h(Arow + kk*32, 64, lane);
#pragma unroll
    for (int nt = 0; nt < 4; ++nt) {
      v16h b = frag_rm_h(Brow + (long)nt*16*64 + kk*32, 64, lane);
      acc[nt] = wmma32(a, b, acc[nt]);
    }
  }
  const int r0 = ibase + ((lane & 16) >> 1);
#pragma unroll
  for (int nt = 0; nt < 4; ++nt) {
    int col = ob64 + nt*16 + (lane & 15);
#pragma unroll
    for (int j = 0; j < 8; ++j) {
      long idx = (long)(r0 + j) * S_ + col;
      Ap[idx] = (f16)((float)Ap[idx] * acc[nt][j]);
    }
  }
}

// ---------------- K7: Y[b,i,head*64+f] = sum_o AT[i,o]*h[o,f]  (final mix)
__global__ void __launch_bounds__(128)
k_mix_final(const f16* __restrict__ AT, const f16* __restrict__ hT,
            f16* __restrict__ Y) {
  const int lane = threadIdx.x & 31, w = threadIdx.x >> 5;
  const int b = blockIdx.z, head = blockIdx.y, bh = b * H_ + head;
  const f16* hTh  = hT + (long)bh * 64 * S_;
  const int ibase = (blockIdx.x * 4 + w) * 16;
  const f16* Arow = AT + ((long)bh * S_ + ibase) * S_;

  v8f acc[4] = {};
  v16h aF0 = frag_rm_h(Arow, S_, lane);
  v16h bF0[4];
#pragma unroll
  for (int nt = 0; nt < 4; ++nt) bF0[nt] = frag_rm_h(hTh + (long)(nt*16)*S_, S_, lane);
  for (int o0 = 0; o0 < S_; o0 += 64) {
    const int oA = o0 + 32;                        // always < S_
    __builtin_prefetch(Arow + (long)(lane & 15) * S_ + oA + 32, 0, 1);
    v16h aF1 = frag_rm_h(Arow + oA, S_, lane);
    v16h bF1[4];
#pragma unroll
    for (int nt = 0; nt < 4; ++nt)
      bF1[nt] = frag_rm_h(hTh + (long)(nt*16)*S_ + oA, S_, lane);
#pragma unroll
    for (int nt = 0; nt < 4; ++nt) acc[nt] = wmma32(aF0, bF0[nt], acc[nt]);
    const int oB = (o0 + 64 < S_) ? o0 + 64 : 0;   // clamped (dead on last iter)
    aF0 = frag_rm_h(Arow + oB, S_, lane);
#pragma unroll
    for (int nt = 0; nt < 4; ++nt)
      bF0[nt] = frag_rm_h(hTh + (long)(nt*16)*S_ + oB, S_, lane);
#pragma unroll
    for (int nt = 0; nt < 4; ++nt) acc[nt] = wmma32(aF1, bF1[nt], acc[nt]);
  }
#pragma unroll
  for (int nt = 0; nt < 4; ++nt) {
    int col = nt*16 + (lane & 15), r0 = (lane & 16) >> 1;
#pragma unroll
    for (int j = 0; j < 8; ++j)
      Y[((long)b*S_ + ibase + r0 + j) * E_ + head*64 + col] = (f16)acc[nt][j];
  }
}

// ---------------- host-side launch ----------------
extern "C" void kernel_launch(void* const* d_in, const int* in_sizes, int n_in,
                              void* d_out, int out_size, void* d_ws, size_t ws_size,
                              hipStream_t stream) {
  (void)in_sizes; (void)n_in; (void)out_size; (void)ws_size;
  const float* x  = (const float*)d_in[0];
  const float* ew = (const float*)d_in[1];
  const float* eb = (const float*)d_in[2];
  const float* nw = (const float*)d_in[3];
  const float* ow = (const float*)d_in[4];
  const float* ob = (const float*)d_in[5];
  float* out = (float*)d_out;

  char* p = (char*)d_ws;
  auto carve = [&](size_t bytes) {
    char* r = p; p += (bytes + 255) & ~(size_t)255; return r;
  };
  f16*   xh    = (f16*)  carve((size_t)TOK * FIN * 2);
  f16*   ewh   = (f16*)  carve((size_t)E_ * FIN * 2);
  f16*   owh   = (f16*)  carve((size_t)FIN * E_ * 2);
  float* xe    = (float*)carve((size_t)TOK * E_ * 4);
  float* inp   = (float*)carve((size_t)TOK * E_ * 4);
  f16*   Wn    = (f16*)  carve(64 * 64 * 2);
  f16*   WnT   = (f16*)  carve(64 * 64 * 2);
  f16*   hT    = (f16*)  carve((size_t)B_ * H_ * 64 * S_ * 2);
  f16*   hrec  = (f16*)  carve((size_t)B_ * H_ * S_ * 64 * 2);
  f16*   areci = (f16*)  carve((size_t)B_ * H_ * S_ * 64 * 2);
  f16*   Y     = (f16*)  carve((size_t)TOK * E_ * 2);
  f16*   AT    = (f16*)  carve((size_t)B_ * H_ * S_ * S_ * 2);

  k_prep_w<<<1, 64, 0, stream>>>(nw, Wn, WnT);
  {
    long n4;
    n4 = (long)TOK * FIN / 4;
    k_cvt_f16<<<(unsigned)((n4 + 255) / 256), 256, 0, stream>>>(x, xh, n4);
    n4 = (long)E_ * FIN / 4;
    k_cvt_f16<<<(unsigned)((n4 + 255) / 256), 256, 0, stream>>>(ew, ewh, n4);
    n4 = (long)FIN * E_ / 4;
    k_cvt_f16<<<(unsigned)((n4 + 255) / 256), 256, 0, stream>>>(ow, owh, n4);
  }
  k_gemm16x64<<<(TOK/16)*(E_/64)/8, 256, 0, stream>>>(xh, ewh, eb, xe, FIN, E_, 1);
  k_l1norm   <<<(TOK*H_ + 255)/256, 256, 0, stream>>>(xe, inp);
  k_nnmf     <<<dim3(TOK/16, H_/4), 128, 0, stream>>>(inp, Wn, WnT, hT, hrec);
  long n8 = (long)B_ * H_ * S_ * S_ / 8;
  k_fill_ones<<<(unsigned)((n8 + 255) / 256), 256, 0, stream>>>(AT, n8);
  for (int it = 0; it < 3; ++it) {
    k_mix_arec<<<dim3(S_/64, H_, B_), 128, 0, stream>>>(AT, hT, WnT, areci);
    k_update  <<<dim3((S_/16)*(S_/64)/8, H_, B_), 256, 0, stream>>>(AT, areci, hrec);
  }
  k_mix_final<<<dim3(S_/64, H_, B_), 128, 0, stream>>>(AT, hT, Y);
  k_gemm16x64<<<(TOK/16)*(FIN/64)/8, 256, 0, stream>>>(Y, owh, ob, out, E_, FIN, 0);
}
